// RelativeDotProductAttentionV2_10642928959688
// MI455X (gfx1250) — matrix operand order repaired and checked by reading, not compiled, not run on hardware
//
#include <hip/hip_runtime.h>

// ---------------------------------------------------------------------------
// Relative dot-product attention (Music Transformer skew), MI455X / gfx1250.
// B=2,H=8,L=2048,D=64, MAX_REL_POS=1024.
// logit[q,k] = q.k + q.embed[1023-(q-k)]  (rel term zero for q-k>1023),
// causal mask, softmax, out = W @ V ; returns (out, W).
// ---------------------------------------------------------------------------

typedef __attribute__((ext_vector_type(16))) _Float16 v16h;
typedef __attribute__((ext_vector_type(8)))  _Float16 v8h;
typedef __attribute__((ext_vector_type(8)))  float    v8f;

#define L_SEQ   2048
#define D_HEAD  64
#define N_BH    16
#define MAXREL  1024

// ---- CDNA5 WMMA fragment loaders (wave32 layouts per CDNA5 ISA) ----
static __device__ __forceinline__ v16h load_a_frag(const _Float16* base, int ld, int lane) {
  const int m = lane & 15, g = lane >> 4;
  const _Float16* p = base + m * ld + g * 8;
  v8h lo = *(const v8h*)(p);
  v8h hi = *(const v8h*)(p + 16);
  v16h r;
#pragma unroll
  for (int i = 0; i < 8; ++i) { r[i] = lo[i]; r[i + 8] = hi[i]; }
  return r;
}
static __device__ __forceinline__ v16h load_b_frag(const _Float16* base, int ld, int lane) {
  const int n = lane & 15, g = lane >> 4;
  const _Float16* p = base + n * ld + g * 8;
  v8h lo = *(const v8h*)(p);
  v8h hi = *(const v8h*)(p + 16);
  v16h r;
#pragma unroll
  for (int i = 0; i < 8; ++i) { r[i] = lo[i]; r[i + 8] = hi[i]; }
  return r;
}
static __device__ __forceinline__ v8f wmma16(v16h a, v16h b, v8f c) {
  return __builtin_amdgcn_wmma_f32_16x16x32_f16(false, a, false, b, (short)0, c,
                                                false, false);
}

// ---- CDNA5 async global->LDS staging (ASYNCcnt path, no VGPR round trip) ----
// Copies 8 KB (one 64x64 f16 tile): 128 threads x 4 x b128.
static __device__ __forceinline__ void async_copy_8k(uint32_t lds_off,
                                                     const _Float16* src, int tid) {
#pragma unroll
  for (int j = 0; j < 4; ++j) {
    const uint64_t ga = (uint64_t)(uintptr_t)src + (uint64_t)(tid + j * 128) * 16u;
    const uint32_t la = lds_off + (uint32_t)(tid + j * 128) * 16u;
    asm volatile("global_load_async_to_lds_b128 %0, %1, off"
                 :: "v"(la), "v"(ga) : "memory");
  }
}
static __device__ __forceinline__ void wait_async0() {
  asm volatile("s_wait_asynccnt 0x0" ::: "memory");
}

// ---------------------------------------------------------------------------
// Precompute kernels: fp32 -> f16 copies of Q,K and a transposed f16 V.
// ---------------------------------------------------------------------------
__global__ void cvt_f16(const float* __restrict__ s, _Float16* __restrict__ d, int n) {
  int i = blockIdx.x * blockDim.x + threadIdx.x;
  const int stride = gridDim.x * blockDim.x;
  for (; i < n; i += stride) d[i] = (_Float16)s[i];
}
__global__ void transpose_v(const float* __restrict__ v, _Float16* __restrict__ vt) {
  // vt[bh][d][k] = v[bh][k][d]
  const int n = N_BH * D_HEAD * L_SEQ;
  int i = blockIdx.x * blockDim.x + threadIdx.x;
  const int stride = gridDim.x * blockDim.x;
  for (; i < n; i += stride) {
    const int bh  = i / (D_HEAD * L_SEQ);
    const int rem = i - bh * (D_HEAD * L_SEQ);
    const int d   = rem / L_SEQ;
    const int k   = rem - d * L_SEQ;
    vt[i] = (_Float16)v[(size_t)bh * L_SEQ * D_HEAD + (size_t)k * D_HEAD + d];
  }
}

// ---------------------------------------------------------------------------
// Main fused kernel. Grid: (32 q-tiles, 16 bh). Block: 128 threads = 4 waves.
// Wave w owns q-rows [q0+16w, q0+16w+16). Dynamic LDS = 144 KB:
//   G[64][1024] f16 (128 KB) | stage[64][64] f16 (8 KB) | Pbuf[64][64] f16 (8 KB)
// ---------------------------------------------------------------------------
__global__ void __launch_bounds__(128)
relattn_main(const _Float16* __restrict__ qh, const _Float16* __restrict__ kh,
             const _Float16* __restrict__ vt, const float* __restrict__ embed,
             float* __restrict__ out, float* __restrict__ wts) {
  extern __shared__ _Float16 smem[];
  _Float16* Glds  = smem;                    // 64 * 1024
  _Float16* stage = smem + 64 * 1024;        // 64 * 64  (K tile / ER chunk / V^T tile)
  _Float16* Pbuf  = stage + 64 * 64;         // 64 * 64
  const uint32_t stage_off = (uint32_t)(uintptr_t)(void*)stage;  // LDS byte offset

  const int bh   = blockIdx.y;
  const int q0   = blockIdx.x * 64;
  const int tid  = threadIdx.x;
  const int w    = tid >> 5;
  const int lane = tid & 31;
  const int lm   = lane & 15, lg = lane >> 4;

  const size_t bh_qk = (size_t)bh * L_SEQ * D_HEAD;

  // Resident Q A-fragments (K=0..31 and K=32..63) for this wave's 16 rows.
  const _Float16* qbase = qh + bh_qk + (size_t)(q0 + 16 * w) * D_HEAD;
  const v16h aq0 = load_a_frag(qbase + 0,  D_HEAD, lane);
  const v16h aq1 = load_a_frag(qbase + 32, D_HEAD, lane);

  // ---- Build G[m][d] = q_m . embed[1023-d] (f16 in LDS) via WMMA ----------
  const int nchunks = min(16, blockIdx.x + 1);
  for (int c = 0; c < nchunks; ++c) {
    __syncthreads();
    for (int e = tid * 4; e < 4096; e += 128 * 4) {           // stage ER chunk (cvt)
      const int r = e >> 6, col = e & 63;                     // ER[dl]=embed[1023-dl]
      const float4 f = *(const float4*)(embed + (size_t)(1023 - (64 * c + r)) * 64 + col);
      stage[r * 64 + col + 0] = (_Float16)f.x;
      stage[r * 64 + col + 1] = (_Float16)f.y;
      stage[r * 64 + col + 2] = (_Float16)f.z;
      stage[r * 64 + col + 3] = (_Float16)f.w;
    }
    __syncthreads();
    v16h bf[8];                                               // hoist all B-frags
#pragma unroll
    for (int sc = 0; sc < 4; ++sc) {
      bf[2 * sc + 0] = load_b_frag(stage + sc * 16 * 64 + 0,  64, lane);
      bf[2 * sc + 1] = load_b_frag(stage + sc * 16 * 64 + 32, 64, lane);
    }
#pragma unroll
    for (int sc = 0; sc < 4; ++sc) {
      v8f g = {};
      g = wmma16(aq0, bf[2 * sc + 0], g);
      g = wmma16(aq1, bf[2 * sc + 1], g);
#pragma unroll
      for (int r = 0; r < 8; ++r)
        Glds[(16 * w + r + 8 * lg) * 1024 + 64 * c + 16 * sc + lm] = (_Float16)g[r];
    }
  }

  const int ktiles = blockIdx.x + 1;       // k0 = 0..q0 covers all causal k
  float mrow[8], srow[8];
#pragma unroll
  for (int r = 0; r < 8; ++r) { mrow[r] = -3.0e38f; srow[r] = 0.0f; }

  // ================= pass A: online softmax statistics =====================
  for (int kt = 0; kt < ktiles; ++kt) {
    const int k0 = kt * 64;
    __syncthreads();
    async_copy_8k(stage_off, kh + bh_qk + (size_t)k0 * D_HEAD, tid);
    if (kt + 1 < ktiles)                                      // global_prefetch_b8
      __builtin_prefetch(kh + bh_qk + (size_t)(k0 + 64) * D_HEAD + tid * 32, 0, 1);
    wait_async0();
    __syncthreads();

    v16h bf[8];
#pragma unroll
    for (int sc = 0; sc < 4; ++sc) {
      bf[2 * sc + 0] = load_b_frag(stage + sc * 16 * 64 + 0,  64, lane);
      bf[2 * sc + 1] = load_b_frag(stage + sc * 16 * 64 + 32, 64, lane);
    }
    v8f s[4];
#pragma unroll
    for (int sc = 0; sc < 4; ++sc) {
      v8f acc = {};
      acc = wmma16(aq0, bf[2 * sc + 0], acc);
      acc = wmma16(aq1, bf[2 * sc + 1], acc);
      s[sc] = acc;
    }
#pragma unroll
    for (int r = 0; r < 8; ++r) {                            // rel + causal mask
      const int row = 16 * w + r + 8 * lg;
      const int q = q0 + row;
#pragma unroll
      for (int sc = 0; sc < 4; ++sc) {
        const int k = k0 + 16 * sc + lm;
        const int d = q - k;
        float v = s[sc][r];
        if (d < 0) v = -3.0e38f;
        else if (d < MAXREL) v += (float)Glds[row * 1024 + d];
        s[sc][r] = v;
      }
    }
#pragma unroll
    for (int r = 0; r < 8; ++r) {                            // online (m,s) update
      float mx = fmaxf(fmaxf(s[0][r], s[1][r]), fmaxf(s[2][r], s[3][r]));
#pragma unroll
      for (int off = 1; off < 16; off <<= 1) mx = fmaxf(mx, __shfl_xor(mx, off, 32));
      const float mnew = fmaxf(mrow[r], mx);
      float ps = __expf(s[0][r] - mnew) + __expf(s[1][r] - mnew) +
                 __expf(s[2][r] - mnew) + __expf(s[3][r] - mnew);
#pragma unroll
      for (int off = 1; off < 16; off <<= 1) ps += __shfl_xor(ps, off, 32);
      srow[r] = srow[r] * __expf(mrow[r] - mnew) + ps;
      mrow[r] = mnew;
    }
  }

  float inv[8];
#pragma unroll
  for (int r = 0; r < 8; ++r) inv[r] = 1.0f / srow[r];

  // ============ pass B: normalized weights out + P @ V =====================
  v8f oacc[4];
#pragma unroll
  for (int sc = 0; sc < 4; ++sc) { v8f z = {}; oacc[sc] = z; }
  float* wbase = wts + (size_t)bh * L_SEQ * L_SEQ;
  const _Float16* vtb = vt + (size_t)bh * D_HEAD * L_SEQ;

  for (int kt = 0; kt < ktiles; ++kt) {
    const int k0 = kt * 64;
    __syncthreads();
    async_copy_8k(stage_off, kh + bh_qk + (size_t)k0 * D_HEAD, tid);
    __builtin_prefetch(vtb + k0 + (size_t)(tid >> 2) * L_SEQ + (tid & 3) * 16, 0, 1);
    wait_async0();
    __syncthreads();

    v16h bf[8];
#pragma unroll
    for (int sc = 0; sc < 4; ++sc) {
      bf[2 * sc + 0] = load_b_frag(stage + sc * 16 * 64 + 0,  64, lane);
      bf[2 * sc + 1] = load_b_frag(stage + sc * 16 * 64 + 32, 64, lane);
    }
    v8f s[4];
#pragma unroll
    for (int sc = 0; sc < 4; ++sc) {
      v8f acc = {};
      acc = wmma16(aq0, bf[2 * sc + 0], acc);
      acc = wmma16(aq1, bf[2 * sc + 1], acc);
      s[sc] = acc;
    }
#pragma unroll
    for (int r = 0; r < 8; ++r) {
      const int row = 16 * w + r + 8 * lg;
      const int q = q0 + row;
#pragma unroll
      for (int sc = 0; sc < 4; ++sc) {
        const int k = k0 + 16 * sc + lm;
        const int d = q - k;
        float v = s[sc][r];
        float p;
        if (d < 0) p = 0.0f;                                  // masked -> exact 0
        else {
          if (d < MAXREL) v += (float)Glds[row * 1024 + d];
          p = __expf(v - mrow[r]) * inv[r];
        }
        __builtin_nontemporal_store(p, &wbase[(size_t)q * L_SEQ + k]);  // streamed
        Pbuf[row * 64 + 16 * sc + lm] = (_Float16)p;          // same-wave LDS
      }
    }
    __syncthreads();                                          // stage free of K
    {                                                         // stage V^T tile [d][kk]
      // 64 rows x 128 B from vt (row stride L); per-lane async b128.
#pragma unroll
      for (int j = 0; j < 4; ++j) {
        const int i = tid + j * 128;                          // i in [0,512)
        const int row = i >> 3, part = i & 7;
        const uint64_t ga = (uint64_t)(uintptr_t)(vtb + (size_t)row * L_SEQ + k0 + part * 8);
        const uint32_t la = stage_off + (uint32_t)i * 16u;
        asm volatile("global_load_async_to_lds_b128 %0, %1, off"
                     :: "v"(la), "v"(ga) : "memory");
      }
      wait_async0();
    }
    __syncthreads();
    const v16h ap0 = load_a_frag(Pbuf + 16 * w * 64 + 0,  64, lane);
    const v16h ap1 = load_a_frag(Pbuf + 16 * w * 64 + 32, 64, lane);
    v16h bv[8];
#pragma unroll
    for (int sc = 0; sc < 4; ++sc) {
      bv[2 * sc + 0] = load_b_frag(stage + sc * 16 * 64 + 0,  64, lane);
      bv[2 * sc + 1] = load_b_frag(stage + sc * 16 * 64 + 32, 64, lane);
    }
#pragma unroll
    for (int sc = 0; sc < 4; ++sc) {
      oacc[sc] = wmma16(ap0, bv[2 * sc + 0], oacc[sc]);
      oacc[sc] = wmma16(ap1, bv[2 * sc + 1], oacc[sc]);
    }
  }

  // ---- write output [bh][q][d] ----
  float* obase = out + bh_qk;
#pragma unroll
  for (int sc = 0; sc < 4; ++sc)
#pragma unroll
    for (int r = 0; r < 8; ++r) {
      const int q = q0 + 16 * w + r + 8 * lg;
      __builtin_nontemporal_store(oacc[sc][r], &obase[(size_t)q * D_HEAD + 16 * sc + lm]);
    }
}

// ---------------------------------------------------------------------------
extern "C" void kernel_launch(void* const* d_in, const int* in_sizes, int n_in,
                              void* d_out, int out_size, void* d_ws, size_t ws_size,
                              hipStream_t stream) {
  (void)in_sizes; (void)n_in; (void)ws_size;
  const float* q     = (const float*)d_in[0];
  const float* k     = (const float*)d_in[1];
  const float* v     = (const float*)d_in[2];
  const float* embed = (const float*)d_in[3];
  // d_in[4] (causal mask) applied analytically.

  const int N = N_BH * L_SEQ * D_HEAD;                 // 2,097,152
  _Float16* qh = (_Float16*)d_ws;                      // 4 MB
  _Float16* kh = qh + N;                               // 4 MB
  _Float16* vt = kh + N;                               // 4 MB (transposed V)

  float* out = (float*)d_out;
  float* wts = out + (size_t)N;                        // weights follow output

  // Zero whole output: provides exact zeros for never-visited k-tiles (k > q0+63).
  hipMemsetAsync(d_out, 0, (size_t)out_size * sizeof(float), stream);

  cvt_f16<<<512, 256, 0, stream>>>(q, qh, N);
  cvt_f16<<<512, 256, 0, stream>>>(k, kh, N);
  transpose_v<<<512, 256, 0, stream>>>(v, vt);

  const size_t smem = (size_t)(64 * 1024 + 64 * 64 + 64 * 64) * sizeof(_Float16); // 147456
  hipFuncSetAttribute((const void*)relattn_main,
                      hipFuncAttributeMaxDynamicSharedMemorySize, (int)smem);
  relattn_main<<<dim3(32, N_BH), 128, smem, stream>>>(qh, kh, vt, embed, out, wts);
}
